// LSTM_2164663517311
// MI455X (gfx1250) — matrix operand order repaired and checked by reading, not compiled, not run on hardware
//
#include <hip/hip_runtime.h>
#include <hip/hip_bf16.h>

typedef __bf16 bf16_t;
typedef __attribute__((ext_vector_type(16))) __bf16 v16bf;
typedef __attribute__((ext_vector_type(8)))  __bf16 v8bf;
typedef __attribute__((ext_vector_type(8)))  float   v8f;

#define BB 64
#define TT 512
#define II 512
#define HH 512
#define NWG 16                    // persistent blocks; 16 x 8 waves = 128 h-tiles
#define GSTRIDE (32*32*32*16)     // per-gate stride in packed W (elements)
#define NSLICE  (32*32*16)        // per-(gate,ntile) contiguous slice (16384 el = 32KB)
#define LDS_BYTES (2*4*NSLICE*2)  // 2 ntiles x 4 gates x 32KB = 256KB

// ---------- convert fp32 X [B,T,I] -> bf16 (same layout) ----------
__global__ void cvt_x_kernel(const float* __restrict__ x, bf16_t* __restrict__ xb) {
  int i = (blockIdx.x * blockDim.x + threadIdx.x) * 4;
  float4 v = *(const float4*)(x + i);
  xb[i + 0] = (bf16_t)v.x;
  xb[i + 1] = (bf16_t)v.y;
  xb[i + 2] = (bf16_t)v.z;
  xb[i + 3] = (bf16_t)v.w;
}

// ---------- pack W_[f,i,o,C] (each [1024,512] f32) into fragment-major bf16 ----------
// Wp[((g*32 + n)*32 + kb)*32 + lane][e] = W_g[(kb*32 + (lane/16)*16 + e)*512 + n*16 + lane%16]
__global__ void pack_w_kernel(const float* __restrict__ Wf, const float* __restrict__ Wi,
                              const float* __restrict__ Wo, const float* __restrict__ Wc,
                              bf16_t* __restrict__ Wp) {
  int idx  = blockIdx.x * blockDim.x + threadIdx.x;   // 0 .. 131071
  int lane = idx & 31;
  int kb   = (idx >> 5)  & 31;
  int n    = (idx >> 10) & 31;
  int g    = (idx >> 15) & 3;
  const float* W = (g == 0) ? Wf : (g == 1) ? Wi : (g == 2) ? Wo : Wc;
  int N     = n * 16 + (lane & 15);
  int kbase = kb * 32 + (lane >> 4) * 16;
  bf16_t* out = Wp + (size_t)idx * 16;
#pragma unroll
  for (int e = 0; e < 16; ++e)
    out[e] = (bf16_t)W[(size_t)(kbase + e) * HH + N];
}

// ---------- persistent LSTM kernel ----------
__global__ __launch_bounds__(256, 1) void lstm_persistent_kernel(
    const bf16_t* __restrict__ Xb,     // [B,T,I] bf16
    const bf16_t* __restrict__ Wp,     // packed weights (global, fragment-major)
    bf16_t* __restrict__ Hbuf,         // 2 x [64,512] bf16, double-buffered
    const float* __restrict__ bfp, const float* __restrict__ bip,
    const float* __restrict__ bop, const float* __restrict__ bcp,
    unsigned* __restrict__ cnt,
    float* __restrict__ out)           // [64,512] f32
{
  extern __shared__ bf16_t sW[];       // [2 ntiles][4 gates][32 kb][32 lanes][16]

  const int lane = threadIdx.x & 31;
  const int wv   = threadIdx.x >> 5;   // 0..7
  const int mtile  = wv & 3;           // 0..3  (batch rows / 16)
  const int ntL    = wv >> 2;          // 0..1  local ntile
  const int ntile  = blockIdx.x * 2 + ntL;   // 0..31 (h cols / 16)
  const int m0 = mtile * 16, n0 = ntile * 16;
  const int ln16 = lane & 15, half = lane >> 4;

  // ---- stage this block's weight slice into LDS (loop-invariant) ----
  for (int c = 0; c < 8; ++c) {
    int cn = c >> 2, cg = c & 3;
    const bf16_t* src = Wp + (size_t)cg * GSTRIDE + (size_t)(blockIdx.x * 2 + cn) * NSLICE;
    bf16_t* dst = sW + (size_t)(cn * 4 + cg) * NSLICE;
    for (int e = threadIdx.x * 8; e < NSLICE; e += 256 * 8)
      *(v8bf*)(dst + e) = *(const v8bf*)(src + e);
  }
  __syncthreads();

  const bf16_t* xrow = Xb + (size_t)(m0 + ln16) * (TT * II);   // this lane's batch row
  const float bF = bfp[n0 + ln16], bI = bip[n0 + ln16];
  const float bO = bop[n0 + ln16], bC = bcp[n0 + ln16];

  const bf16_t* wlds = sW + (size_t)ntL * (4 * NSLICE) + (size_t)lane * 16;

  auto ldfrag = [](const bf16_t* p0, const bf16_t* p1) -> v16bf {
    v8bf lo = *(const v8bf*)p0;
    v8bf hi = *(const v8bf*)p1;
    v16bf r;
#pragma unroll
    for (int e = 0; e < 8; ++e) { r[e] = lo[e]; r[8 + e] = hi[e]; }
    return r;
  };

  v8f cst = {};                        // c-state tile lives in registers all T steps
  for (int t = 0; t < TT; ++t) {
    const bf16_t* hread  = Hbuf + (size_t)(t & 1) * (BB * HH);
    bf16_t*       hwrite = Hbuf + (size_t)((t + 1) & 1) * (BB * HH);
    const bf16_t* hrow = hread + (size_t)(m0 + ln16) * HH;
    const bf16_t* xrt  = xrow + (size_t)t * II;

    auto ldA = [&](int kb) -> v16bf {
      const bf16_t* ar = (kb < 16) ? (xrt + kb * 32) : (hrow + (kb - 16) * 32);
      return ldfrag(ar + half * 16, ar + half * 16 + 8);
    };
    auto ldB = [&](int kb, int g) -> v16bf {
      const bf16_t* wfp = wlds + (size_t)kb * (32 * 16) + (size_t)g * NSLICE;
      return ldfrag(wfp, wfp + 8);
    };

    // bias folded into accumulator init (bias is per-column == per-lane constant)
    v8f acc0, acc1, acc2, acc3;
#pragma unroll
    for (int r = 0; r < 8; ++r) { acc0[r] = bF; acc1[r] = bI; acc2[r] = bO; acc3[r] = bC; }

    // ---- software-pipelined K loop: loads for kb+1 are pinned ABOVE the WMMAs
    //      of kb with a scheduling barrier, so WMMA never waits on fresh loads ----
    v16bf a  = ldA(0);
    v16bf b0 = ldB(0, 0), b1 = ldB(0, 1), b2 = ldB(0, 2), b3 = ldB(0, 3);
#pragma unroll
    for (int kb = 0; kb < 32; ++kb) {
      v16bf an = a, b0n = b0, b1n = b1, b2n = b2, b3n = b3;
      if (kb < 31) {
        an  = ldA(kb + 1);
        b0n = ldB(kb + 1, 0);
        b1n = ldB(kb + 1, 1);
        b2n = ldB(kb + 1, 2);
        b3n = ldB(kb + 1, 3);
      }
      __builtin_amdgcn_sched_barrier(0);   // keep prefetch loads above the WMMAs
      acc0 = __builtin_amdgcn_wmma_f32_16x16x32_bf16(false, a, false, b0, (short)0, acc0, false, false);
      acc1 = __builtin_amdgcn_wmma_f32_16x16x32_bf16(false, a, false, b1, (short)0, acc1, false, false);
      acc2 = __builtin_amdgcn_wmma_f32_16x16x32_bf16(false, a, false, b2, (short)0, acc2, false, false);
      acc3 = __builtin_amdgcn_wmma_f32_16x16x32_bf16(false, a, false, b3, (short)0, acc3, false, false);
      a = an; b0 = b0n; b1 = b1n; b2 = b2n; b3 = b3n;
    }

    // gates + state update (fp32); C/D layout: row = m0 + half*8 + r, col = n0 + ln16
#pragma unroll
    for (int r = 0; r < 8; ++r) {
      float fv = 1.f / (1.f + __expf(-acc0[r]));
      float iv = 1.f / (1.f + __expf(-acc1[r]));
      float ov = 1.f / (1.f + __expf(-acc2[r]));
      float cg = 2.f / (1.f + __expf(-2.f * acc3[r])) - 1.f;   // tanh
      float cn = fv * cst[r] + iv * cg;
      cst[r] = cn;
      float th = 2.f / (1.f + __expf(-2.f * cn)) - 1.f;        // tanh
      float hv = ov * th;
      int row = m0 + half * 8 + r;
      hwrite[(size_t)row * HH + n0 + ln16] = (bf16_t)hv;
      if (t == TT - 1) out[(size_t)row * HH + n0 + ln16] = hv;
    }

    // grid-wide sync (monotonic counter), one per step
    __threadfence();
    __syncthreads();
    if (threadIdx.x == 0) {
      atomicAdd(cnt, 1u);
      unsigned target = (unsigned)(t + 1) * NWG;
      volatile unsigned* vc = (volatile unsigned*)cnt;
      while (*vc < target) __builtin_amdgcn_s_sleep(8);
    }
    __syncthreads();
    __threadfence();
  }
}

extern "C" void kernel_launch(void* const* d_in, const int* in_sizes, int n_in,
                              void* d_out, int out_size, void* d_ws, size_t ws_size,
                              hipStream_t stream) {
  const float* x  = (const float*)d_in[0];   // [64,512,512]
  const float* Wf = (const float*)d_in[1];
  const float* bf = (const float*)d_in[2];
  const float* Wi = (const float*)d_in[3];
  const float* bi = (const float*)d_in[4];
  const float* Wo = (const float*)d_in[5];
  const float* bo = (const float*)d_in[6];
  const float* Wc = (const float*)d_in[7];
  const float* bc = (const float*)d_in[8];

  char* ws = (char*)d_ws;
  unsigned* cnt = (unsigned*)ws;                              // 256 B
  bf16_t* Hbuf  = (bf16_t*)(ws + 256);                        // 2*64*512*2  = 128 KB
  bf16_t* Wp    = (bf16_t*)(ws + 256 + 131072);               // 4 MB
  bf16_t* Xb    = (bf16_t*)(ws + 256 + 131072 + 4194304);     // 32 MB

  hipMemsetAsync(cnt, 0, 256, stream);
  hipMemsetAsync(Hbuf, 0, 131072, stream);

  cvt_x_kernel<<<16384, 256, 0, stream>>>(x, Xb);                  // 64*512*512 / (256*4)
  pack_w_kernel<<<512, 256, 0, stream>>>(Wf, Wi, Wo, Wc, Wp);      // 131072 fragments
  lstm_persistent_kernel<<<NWG, 256, LDS_BYTES, stream>>>(Xb, Wp, Hbuf, bf, bi, bo, bc,
                                                          cnt, (float*)d_out);
}
